// WindowAttentionConv_65146063946592
// MI455X (gfx1250) — matrix operand order repaired and checked by reading, not compiled, not run on hardware
//
#include <hip/hip_runtime.h>
#include <hip/hip_bf16.h>

// MI455X (gfx1250) Swin window-attention. f16 WMMA (f32 accum) for all three
// GEMM stages; fp32 softmax/bias/mask. One wave32 = one 16-row WMMA stripe.

typedef __attribute__((ext_vector_type(16))) _Float16 v16h;
typedef __attribute__((ext_vector_type(8)))  _Float16 v8h;
typedef __attribute__((ext_vector_type(8)))  float    v8f;

#define WMMA_F16(a, b, c) \
  __builtin_amdgcn_wmma_f32_16x16x32_f16(false, (a), false, (b), (short)0, (c), false, false)

static __device__ __forceinline__ v16h cat8(v8h lo, v8h hi) {
  return __builtin_shufflevector(lo, hi, 0,1,2,3,4,5,6,7,8,9,10,11,12,13,14,15);
}

// ---------------------------------------------------------------- constants
// B_=2048, N=49 (pad 64), C=384, H=12, hd=32, nW=256
#define NROWS   100352          // 2048*49  (== 1568*64, exact tiling)
#define NPAD    64
#define SCALE   0.17677669529663689f  // 32^-0.5

// ---------------------------------------------------------------- kernel 1
__global__ __launch_bounds__(256) void cvt_weights(
    const float* __restrict__ wq, const float* __restrict__ wp,
    _Float16* __restrict__ wq16, _Float16* __restrict__ wp16) {
  int i = blockIdx.x * 256 + threadIdx.x;
  if (i < 384 * 1152) wq16[i] = (_Float16)wq[i];
  if (i < 384 * 384)  wp16[i] = (_Float16)wp[i];
}

// ---------------------------------------------------------------- kernel 2
// qkv = x @ Wqkv + b ; scatter to q (scaled, row-major), kT (transposed), v.
__global__ __launch_bounds__(128) void qkv_gemm(
    const float* __restrict__ x, const _Float16* __restrict__ W16,
    const float* __restrict__ bqkv,
    _Float16* __restrict__ qb, _Float16* __restrict__ kT,
    _Float16* __restrict__ vb) {
  const int lane = threadIdx.x & 31;
  const int wave = threadIdx.x >> 5;
  const int lrow = lane & 15;
  const int half = lane >> 4;
  const int m0   = blockIdx.x * 64 + wave * 16;   // wave's row base
  const int row  = m0 + lrow;                     // A-frag row for this lane

  // Cache all 12 K-step A fragments (x read exactly once, cvt fp32->f16).
  v16h A[12];
  const float* xr = x + (size_t)row * 384;
  const int c0 = half ? 8 : 0;
  const int c1 = c0 + 16;
#pragma unroll
  for (int k = 0; k < 12; ++k) {
    v16h a;
#pragma unroll
    for (int j = 0; j < 8; ++j) {
      a[j]     = (_Float16)xr[k * 32 + c0 + j];
      a[8 + j] = (_Float16)xr[k * 32 + c1 + j];
    }
    A[k] = a;
  }

#pragma unroll 1
  for (int nt = 0; nt < 18; ++nt) {               // 18 * 64 = 1152 cols
    v8f acc[4] = {};
#pragma unroll
    for (int k = 0; k < 12; ++k) {
#pragma unroll
      for (int sub = 0; sub < 4; ++sub) {
        v16h b;
#pragma unroll
        for (int j = 0; j < 8; ++j) {
          b[j]     = W16[(size_t)(k * 32 + c0 + j) * 1152 + nt * 64 + sub * 16 + lrow];
          b[8 + j] = W16[(size_t)(k * 32 + c1 + j) * 1152 + nt * 64 + sub * 16 + lrow];
        }
        acc[sub] = WMMA_F16(A[k], b, acc[sub]);
      }
    }
    // scatter-store C fragments into q / kT / v f16 buffers
#pragma unroll
    for (int sub = 0; sub < 4; ++sub) {
      const int c     = nt * 64 + sub * 16 + lrow;  // output column 0..1151
      const int three = c / 384;                    // 0=q 1=k 2=v (tile-uniform)
      const int cr    = c - three * 384;
      const int h     = cr >> 5;
      const int d     = cr & 31;
      const float bias = bqkv[c];
#pragma unroll
      for (int r = 0; r < 8; ++r) {
        const int R = m0 + r + half * 8;            // global row (b*49+n)
        const int b_ = (unsigned)R / 49u;
        const int n  = R - b_ * 49;
        const size_t bh = (size_t)b_ * 12 + h;
        const float v = acc[sub][r] + bias;
        if (three == 0)      qb[(bh * NPAD + n) * 32 + d] = (_Float16)(v * SCALE);
        else if (three == 1) kT[(bh * 32 + d) * NPAD + n] = (_Float16)v;
        else                 vb[(bh * NPAD + n) * 32 + d] = (_Float16)v;
      }
    }
  }
}

// ---------------------------------------------------------------- kernel 3
// Per (b,h): S = q·kT (+bias+mask), softmax -> attn out (fp32), O = P·V (f16).
__global__ __launch_bounds__(128) void attn_kernel(
    const _Float16* __restrict__ qb, const _Float16* __restrict__ kT,
    const _Float16* __restrict__ vb, const float* __restrict__ bias_table,
    const float* __restrict__ mask, float* __restrict__ attn_out,
    _Float16* __restrict__ ob) {
  __shared__ __align__(16) _Float16 lp[4][16][80];  // per-wave P stripe, f16

  const int lane = threadIdx.x & 31;
  const int wave = threadIdx.x >> 5;
  const int lrow = lane & 15;
  const int half = lane >> 4;
  const int bh   = blockIdx.x;                  // b*12 + h
  const int b_   = (unsigned)bh / 12u;
  const int h    = bh - b_ * 12;
  const int w    = b_ & 255;                    // window mask id = b % nW
  const size_t qkbase = (size_t)bh * (NPAD * 32);

  // ---- A fragment: 16 q rows (hd=32 == exactly one f16-WMMA K step)
  const int c0 = half ? 8 : 0;
  const int c1 = c0 + 16;
  const _Float16* qr = qb + qkbase + (size_t)(wave * 16 + lrow) * 32;
  const v16h aq = cat8(*(const v8h*)(qr + c0), *(const v8h*)(qr + c1));

  // ---- S = q · k^T  (4 column tiles of 16 keys)
  v8f S[4];
#pragma unroll
  for (int nt = 0; nt < 4; ++nt) {
    v16h bk;
#pragma unroll
    for (int j = 0; j < 8; ++j) {
      bk[j]     = kT[qkbase + (size_t)(c0 + j) * NPAD + nt * 16 + lrow];
      bk[8 + j] = kT[qkbase + (size_t)(c1 + j) * NPAD + nt * 16 + lrow];
    }
    v8f z = {};
    S[nt] = WMMA_F16(aq, bk, z);
  }

  // ---- relative-position bias + window mask; -inf on padded key columns
#pragma unroll
  for (int nt = 0; nt < 4; ++nt) {
    const int m   = nt * 16 + lrow;
    const bool ok = m < 49;
    const int im  = (unsigned)m / 7u;
    const int jm  = m - im * 7;
#pragma unroll
    for (int r = 0; r < 8; ++r) {
      const int n  = wave * 16 + r + half * 8;
      const int nc = n < 49 ? n : 48;           // clamp pad rows (discarded)
      if (ok) {
        const int in_ = (unsigned)nc / 7u;
        const int jn  = nc - in_ * 7;
        const int idx = (in_ - im + 6) * 13 + (jn - jm + 6);
        S[nt][r] += bias_table[idx * 12 + h]
                  + mask[(size_t)w * 2401 + nc * 49 + m];
      } else {
        S[nt][r] = -1e30f;
      }
    }
  }

  // ---- row softmax: reduce across 16 lanes of each half (wave32 shuffles)
#pragma unroll
  for (int r = 0; r < 8; ++r) {
    float mx = fmaxf(fmaxf(S[0][r], S[1][r]), fmaxf(S[2][r], S[3][r]));
    mx = fmaxf(mx, __shfl_xor(mx, 1));
    mx = fmaxf(mx, __shfl_xor(mx, 2));
    mx = fmaxf(mx, __shfl_xor(mx, 4));
    mx = fmaxf(mx, __shfl_xor(mx, 8));
    float s = 0.f;
#pragma unroll
    for (int f = 0; f < 4; ++f) {
      const float e = __expf(S[f][r] - mx);
      S[f][r] = e;
      s += e;
    }
    s += __shfl_xor(s, 1);
    s += __shfl_xor(s, 2);
    s += __shfl_xor(s, 4);
    s += __shfl_xor(s, 8);
    const float is = 1.f / s;
#pragma unroll
    for (int f = 0; f < 4; ++f) S[f][r] *= is;
  }

  // ---- store attn probs (fp32, output #2) and P stripe to LDS (f16)
  const size_t abase = (size_t)bh * 2401;
#pragma unroll
  for (int nt = 0; nt < 4; ++nt) {
    const int m = nt * 16 + lrow;
#pragma unroll
    for (int r = 0; r < 8; ++r) {
      const int n = wave * 16 + r + half * 8;
      if (n < 49 && m < 49)
        attn_out[abase + (size_t)n * 49 + m] = S[nt][r];
      lp[wave][r + half * 8][m] = (_Float16)S[nt][r];
    }
  }
  __syncthreads();

  // ---- O = P · V   (K = 64 padded keys -> 2 K steps; N = hd = 32 -> 2 tiles)
  v8f O[2] = {};
#pragma unroll
  for (int ks = 0; ks < 2; ++ks) {
    const int kb = ks * 32;
    const _Float16* pr = &lp[wave][lrow][0];
    const v16h ap = cat8(*(const v8h*)(pr + kb + c0), *(const v8h*)(pr + kb + c1));
#pragma unroll
    for (int sub = 0; sub < 2; ++sub) {
      v16h bv;
#pragma unroll
      for (int j = 0; j < 8; ++j) {
        bv[j]     = vb[qkbase + (size_t)(kb + c0 + j) * 32 + sub * 16 + lrow];
        bv[8 + j] = vb[qkbase + (size_t)(kb + c1 + j) * 32 + sub * 16 + lrow];
      }
      O[sub] = WMMA_F16(ap, bv, O[sub]);
    }
  }
  // ---- store O (f16) in [b*49+n][h*32+d] layout for the projection GEMM
#pragma unroll
  for (int sub = 0; sub < 2; ++sub) {
#pragma unroll
    for (int r = 0; r < 8; ++r) {
      const int n = wave * 16 + r + half * 8;
      if (n < 49)
        ob[((size_t)b_ * 49 + n) * 384 + h * 32 + sub * 16 + lrow] = (_Float16)O[sub][r];
    }
  }
}

// ---------------------------------------------------------------- kernel 4
// out = O @ Wproj + bproj  (fp32 result straight to d_out)
__global__ __launch_bounds__(128) void proj_gemm(
    const _Float16* __restrict__ ob, const _Float16* __restrict__ W16,
    const float* __restrict__ bproj, float* __restrict__ out) {
  const int lane = threadIdx.x & 31;
  const int wave = threadIdx.x >> 5;
  const int lrow = lane & 15;
  const int half = lane >> 4;
  const int m0   = blockIdx.x * 64 + wave * 16;
  const int row  = m0 + lrow;
  const int c0 = half ? 8 : 0;
  const int c1 = c0 + 16;

  v16h A[12];
  const _Float16* orow = ob + (size_t)row * 384;
#pragma unroll
  for (int k = 0; k < 12; ++k)
    A[k] = cat8(*(const v8h*)(orow + k * 32 + c0), *(const v8h*)(orow + k * 32 + c1));

#pragma unroll 1
  for (int nt = 0; nt < 6; ++nt) {                // 6 * 64 = 384 cols
    v8f acc[4] = {};
#pragma unroll
    for (int k = 0; k < 12; ++k) {
#pragma unroll
      for (int sub = 0; sub < 4; ++sub) {
        v16h b;
#pragma unroll
        for (int j = 0; j < 8; ++j) {
          b[j]     = W16[(size_t)(k * 32 + c0 + j) * 384 + nt * 64 + sub * 16 + lrow];
          b[8 + j] = W16[(size_t)(k * 32 + c1 + j) * 384 + nt * 64 + sub * 16 + lrow];
        }
        acc[sub] = WMMA_F16(A[k], b, acc[sub]);
      }
    }
#pragma unroll
    for (int sub = 0; sub < 4; ++sub) {
      const int c = nt * 64 + sub * 16 + lrow;
      const float bias = bproj[c];
#pragma unroll
      for (int r = 0; r < 8; ++r) {
        const int R = m0 + r + half * 8;
        out[(size_t)R * 384 + c] = acc[sub][r] + bias;
      }
    }
  }
}

// ---------------------------------------------------------------- launcher
extern "C" void kernel_launch(void* const* d_in, const int* in_sizes, int n_in,
                              void* d_out, int out_size, void* d_ws, size_t ws_size,
                              hipStream_t stream) {
  const float* x     = (const float*)d_in[0];
  // d_in[1] = input_shape (unused by attention path)
  const float* mask  = (const float*)d_in[2];
  const float* Wqkv  = (const float*)d_in[3];
  const float* bqkv  = (const float*)d_in[4];
  const float* btab  = (const float*)d_in[5];
  const float* Wproj = (const float*)d_in[6];
  const float* bproj = (const float*)d_in[7];

  float* out  = (float*)d_out;
  float* attn = out + (size_t)NROWS * 384;        // tuple output #2

  char* ws = (char*)d_ws;
  _Float16* wq16 = (_Float16*)ws;  ws += (size_t)384 * 1152 * 2;
  _Float16* wp16 = (_Float16*)ws;  ws += (size_t)384 * 384  * 2;
  _Float16* qb   = (_Float16*)ws;  ws += (size_t)2048 * 12 * NPAD * 32 * 2;
  _Float16* kT   = (_Float16*)ws;  ws += (size_t)2048 * 12 * 32 * NPAD * 2;
  _Float16* vb   = (_Float16*)ws;  ws += (size_t)2048 * 12 * NPAD * 32 * 2;
  _Float16* ob   = (_Float16*)ws;  ws += (size_t)NROWS * 384 * 2;

  cvt_weights<<<dim3((384 * 1152 + 255) / 256), dim3(256), 0, stream>>>(
      Wqkv, Wproj, wq16, wp16);
  qkv_gemm<<<dim3(NROWS / 64), dim3(128), 0, stream>>>(
      x, wq16, bqkv, qb, kT, vb);
  attn_kernel<<<dim3(2048 * 12), dim3(128), 0, stream>>>(
      qb, kT, vb, btab, mask, attn, ob);
  proj_gemm<<<dim3(NROWS / 64), dim3(128), 0, stream>>>(
      ob, wp16, bproj, out);
}
